// Seq2SeqModel_33758442946609
// MI455X (gfx1250) — matrix operand order repaired and checked by reading, not compiled, not run on hardware
//
#include <hip/hip_runtime.h>
#include <hip/hip_bf16.h>
#include <math.h>

#define V_   10000
#define H_   512
#define L_   2
#define B_   32
#define T_   128
#define HD_  256
#define BT_  (B_*T_)

typedef _Float16 h16;
typedef h16   v8h  __attribute__((ext_vector_type(8)));
typedef h16   v16h __attribute__((ext_vector_type(16)));
typedef float v8f  __attribute__((ext_vector_type(8)));

// ---------------- WMMA fragment helpers (CDNA5 16x16x32 f16, wave32) -------
__device__ __forceinline__ v16h load_a16(const h16* A, size_t lda, int m0, int k0, int lane) {
  int m  = m0 + (lane & 15);
  int hh = lane >> 4;
  const h16* p = A + (size_t)m * lda + (size_t)(k0 + 8*hh);
  v8h lo = *(const v8h*)p;
  v8h hi = *(const v8h*)(p + 16);
  v16h r;
#pragma unroll
  for (int i = 0; i < 8; i++) { r[i] = lo[i]; r[i+8] = hi[i]; }
  return r;
}

__device__ __forceinline__ v16h load_b16(const h16* Bt, size_t ldb, int n0, int k0, int lane) {
  int n  = n0 + (lane & 15);
  int hh = lane >> 4;
  const h16* p = Bt + (size_t)n * ldb + (size_t)(k0 + 8*hh);
  v8h b0 = *(const v8h*)p;
  v8h b1 = *(const v8h*)(p + 16);
  v16h r;
#pragma unroll
  for (int i = 0; i < 8; i++) { r[i] = b0[i]; r[i+8] = b1[i]; }
  return r;
}

#define WMMA_F32_F16(a, b, c) \
  __builtin_amdgcn_wmma_f32_16x16x32_f16(false, (a), false, (b), (short)0, (c), false, false)

__device__ __forceinline__ float sigm(float x) { return 1.f / (1.f + __expf(-x)); }

// async global->LDS b128 copy (CDNA5, tracked by ASYNCcnt)
__device__ __forceinline__ void async_ld_b128(unsigned lds_off, const void* gaddr) {
  asm volatile("global_load_async_to_lds_b128 %0, %1, off"
               :: "v"(lds_off), "v"((unsigned long long)gaddr) : "memory");
}
__device__ __forceinline__ unsigned lds_off32(const void* p) {
  return (unsigned)(unsigned long long)p;   // generic LDS ptr low 32 = LDS offset
}

// ---------------- LDS-tiled WMMA GEMM, async double-buffered ---------------
// 256 threads = 8 waves; 128x128 C tile per WG; per 32-K step the next
// 128x32 A / 128x32 Bt panels stream into LDS via async b128 loads while the
// current panel feeds 8 WMMAs per wave (64x32 register tile).
// Requires M%128==0, N%128==0, K%32==0. grid=(N/128, M/128, batch).
__global__ __launch_bounds__(256) void gemm_t128(
    const h16* __restrict__ A,  long lda,  long sA,
    const h16* __restrict__ Bt, long ldb,  long sB,
    float* __restrict__ C,      long ldc,  long sC,
    h16*  __restrict__ Ch,      long ldch, long sCh,
    const float* __restrict__ bias0, const float* __restrict__ bias1,
    int K, int relu)
{
  __shared__ __align__(16) h16 sAt[2][128 * 32];   // 2 x 8 KB
  __shared__ __align__(16) h16 sBt[2][128 * 32];   // 2 x 8 KB
  const int tid  = threadIdx.x;
  const int lane = tid & 31;
  const int wave = tid >> 5;          // 0..7
  const int wm   = wave & 1;          // 2 groups of 64 rows
  const int wn   = wave >> 1;         // 4 groups of 32 cols
  const long blockN = (long)blockIdx.x * 128;
  const long blockM = (long)blockIdx.y * 128;
  const size_t zb = blockIdx.z;
  A  += zb * (size_t)sA;
  Bt += zb * (size_t)sB;

  // copy mapping: 128 rows x 4 v8h chunks; thread covers rows r0 and r0+64
  const int r0 = tid >> 2;
  const int c0 = (tid & 3) * 8;
  const h16* gA0 = A  + (size_t)(blockM + r0)      * lda + c0;
  const h16* gA1 = A  + (size_t)(blockM + r0 + 64) * lda + c0;
  const h16* gB0 = Bt + (size_t)(blockN + r0)      * ldb + c0;
  const h16* gB1 = Bt + (size_t)(blockN + r0 + 64) * ldb + c0;
  unsigned lA0[2], lA1[2], lB0[2], lB1[2];
#pragma unroll
  for (int u = 0; u < 2; u++) {
    lA0[u] = lds_off32(&sAt[u][r0 * 32 + c0]);
    lA1[u] = lds_off32(&sAt[u][(r0 + 64) * 32 + c0]);
    lB0[u] = lds_off32(&sBt[u][r0 * 32 + c0]);
    lB1[u] = lds_off32(&sBt[u][(r0 + 64) * 32 + c0]);
  }

  v16h afr[4];
  v16h bfr[2];
  v8f acc[4][2];
#pragma unroll
  for (int i = 0; i < 4; i++)
#pragma unroll
    for (int j = 0; j < 2; j++) acc[i][j] = (v8f){};

  // prologue: stream panel 0 into buffer 0
  async_ld_b128(lA0[0], gA0);
  async_ld_b128(lA1[0], gA1);
  async_ld_b128(lB0[0], gB0);
  async_ld_b128(lB1[0], gB1);

  for (int k0 = 0; k0 < K; k0 += 32) {
    const int cur = (k0 >> 5) & 1;
    if (k0 + 32 < K) {               // stream next panel into other buffer
      const int nxt = cur ^ 1;
      async_ld_b128(lA0[nxt], gA0 + k0 + 32);
      async_ld_b128(lA1[nxt], gA1 + k0 + 32);
      async_ld_b128(lB0[nxt], gB0 + k0 + 32);
      async_ld_b128(lB1[nxt], gB1 + k0 + 32);
      asm volatile("s_wait_asynccnt 0x4" ::: "memory");  // cur panel complete
    } else {
      asm volatile("s_wait_asynccnt 0x0" ::: "memory");
    }
    __syncthreads();                 // all waves' cur panels visible
#pragma unroll
    for (int tm = 0; tm < 4; tm++)
      afr[tm] = load_a16(sAt[cur], 32, wm * 64 + tm * 16, 0, lane);
#pragma unroll
    for (int tn = 0; tn < 2; tn++)
      bfr[tn] = load_b16(sBt[cur], 32, wn * 32 + tn * 16, 0, lane);
#pragma unroll
    for (int tn = 0; tn < 2; tn++)
#pragma unroll
      for (int tm = 0; tm < 4; tm++)
        acc[tm][tn] = WMMA_F32_F16(afr[tm], bfr[tn], acc[tm][tn]);
    __syncthreads();                 // protect cur before it is re-streamed
  }

  const int nn = lane & 15;
  const int hh = lane >> 4;
#pragma unroll
  for (int tn = 0; tn < 2; tn++) {
    long n = blockN + wn * 32 + tn * 16 + nn;
    float bv = 0.f;
    if (bias0) bv += bias0[n];
    if (bias1) bv += bias1[n];
#pragma unroll
    for (int tm = 0; tm < 4; tm++) {
#pragma unroll
      for (int r = 0; r < 8; r++) {
        long m = blockM + wm * 64 + tm * 16 + 8 * hh + r;
        float v = acc[tm][tn][r] + bv;
        if (relu) v = fmaxf(v, 0.f);
        if (C)  C [zb * (size_t)sC  + (size_t)m * ldc  + n] = v;
        if (Ch) Ch[zb * (size_t)sCh + (size_t)m * ldch + n] = (h16)v;
      }
    }
  }
}

// ---------------- 16x16-per-wave WMMA GEMM (edge/tail path) ----------------
__global__ __launch_bounds__(32) void gemm16(
    const h16* __restrict__ A,  long lda,  long sA,
    const h16* __restrict__ Bt, long ldb,  long sB,
    float* __restrict__ C,      long ldc,  long sC,
    h16*  __restrict__ Ch,      long ldch, long sCh,
    const float* __restrict__ bias0, const float* __restrict__ bias1,
    int K, int relu)
{
  const int lane = threadIdx.x;
  const int n0 = blockIdx.x * 16;
  const int m0 = blockIdx.y * 16;
  const size_t zb = blockIdx.z;
  A  += zb * (size_t)sA;
  Bt += zb * (size_t)sB;
  v8f acc = {};
  for (int k0 = 0; k0 < K; k0 += 32) {
    v16h a = load_a16(A,  (size_t)lda, m0, k0, lane);
    v16h b = load_b16(Bt, (size_t)ldb, n0, k0, lane);
    acc = WMMA_F32_F16(a, b, acc);
  }
  const int n  = n0 + (lane & 15);
  const int hh = lane >> 4;
  float bv = 0.f;
  if (bias0) bv += bias0[n];
  if (bias1) bv += bias1[n];
#pragma unroll
  for (int r = 0; r < 8; r++) {
    int m = m0 + 8*hh + r;
    float v = acc[r] + bv;
    if (relu) v = fmaxf(v, 0.f);
    if (C)  C [zb*(size_t)sC  + (size_t)m*ldc  + n] = v;
    if (Ch) Ch[zb*(size_t)sCh + (size_t)m*ldch + n] = (h16)v;
  }
}

// ---------------- elementwise kernels --------------------------------------
__global__ void f32_to_f16_k(const float* __restrict__ s, h16* __restrict__ d, long n) {
  long i = (long)blockIdx.x * blockDim.x + threadIdx.x;
  long st = (long)gridDim.x * blockDim.x;
  for (; i < n; i += st) d[i] = (h16)s[i];
}

__global__ void embed_k(const int* __restrict__ tok, const float* __restrict__ Wemb,
                        const float* __restrict__ bemb, h16* __restrict__ out) {
  long i = (long)blockIdx.x * blockDim.x + threadIdx.x;
  if (i >= (long)BT_ * H_) return;
  int hh  = (int)(i & (H_ - 1));
  long row = i >> 9;              // t*B + b
  int b = (int)(row & (B_ - 1));
  int t = (int)(row >> 5);
  int id = tok[b * T_ + t];
  out[i] = (h16)(Wemb[(size_t)hh * V_ + id] + bemb[hh]);
}

__global__ void transpose_k(const h16* __restrict__ out, h16* __restrict__ outT) {
  long i = (long)blockIdx.x * blockDim.x + threadIdx.x;
  if (i >= (long)B_ * H_ * T_) return;
  int t = (int)(i & (T_ - 1));
  long r = i >> 7;
  int dd = (int)(r & (H_ - 1));
  int b  = (int)(r >> 9);
  outT[i] = out[((size_t)t * B_ + b) * H_ + dd];
}

__global__ void concat_fo_k(const h16* __restrict__ fo, h16* __restrict__ info) {
  long i = (long)blockIdx.x * blockDim.x + threadIdx.x;
  if (i >= (long)BT_ * H_) return;
  int hh = (int)(i & (H_ - 1));
  long r = i >> 9;
  int t = (int)(r & (T_ - 1));
  int b = (int)(r >> 7);
  info[((size_t)(b * T_ + t)) * (2 * H_) + H_ + hh] = fo[((size_t)t * B_ + b) * H_ + hh];
}

__global__ __launch_bounds__(128) void softmax_k(const float* __restrict__ S,
                                                 h16* __restrict__ att) {
  __shared__ float red[T_];
  const long row = blockIdx.x;          // b*T + i
  const int j = threadIdx.x;
  float v = S[row * T_ + j];
  red[j] = v; __syncthreads();
  for (int off = 64; off > 0; off >>= 1) {
    if (j < off) red[j] = fmaxf(red[j], red[j + off]);
    __syncthreads();
  }
  float mx = red[0]; __syncthreads();
  float e = __expf(v - mx);
  red[j] = e; __syncthreads();
  for (int off = 64; off > 0; off >>= 1) {
    if (j < off) red[j] += red[j + off];
    __syncthreads();
  }
  att[row * T_ + j] = (h16)(e / red[0]);
}

// ---------------- encoder LSTM scan (one WG per direction) -----------------
__global__ __launch_bounds__(1024) void enc_scan_k(
    const float* __restrict__ xg0, const float* __restrict__ xg1,
    const h16* __restrict__ whh_l,
    h16* __restrict__ xnext,
    float* __restrict__ gsc,
    float* __restrict__ fh, float* __restrict__ fc,
    int l)
{
  const int d = blockIdx.x;
  const float* xg = d ? xg1 : xg0;
  const h16* whh = whh_l + (size_t)d * (4 * HD_) * HD_;
  float* g = gsc + (size_t)d * B_ * (4 * HD_);
  const size_t foff = (size_t)(l * 2 + d) * B_ * HD_;

  __shared__ __align__(16) h16 hbuf[B_ * HD_];   // 16 KB
  const int tid  = threadIdx.x;
  const int lane = tid & 31;
  const int wave = tid >> 5;

  float c[8];
#pragma unroll
  for (int j = 0; j < 8; j++) c[j] = 0.f;
  for (int i = tid; i < B_ * HD_; i += 1024) hbuf[i] = (h16)0.f;
  __syncthreads();

  for (int s = 0; s < T_; s++) {
    const int t = d ? (T_ - 1 - s) : s;
    for (int tile = wave; tile < 128; tile += 32) {
      const int m0 = (tile & 1) * 16;
      const int n0 = (tile >> 1) * 16;
      v8f acc = {};
      for (int k0 = 0; k0 < HD_; k0 += 32) {
        v16h a = load_a16(hbuf, HD_, m0, k0, lane);
        v16h b = load_b16(whh,  HD_, n0, k0, lane);
        acc = WMMA_F32_F16(a, b, acc);
      }
      const int n  = n0 + (lane & 15);
      const int hh = lane >> 4;
#pragma unroll
      for (int r = 0; r < 8; r++)
        g[(size_t)(m0 + 8 * hh + r) * (4 * HD_) + n] = acc[r];
    }
    __threadfence();
    __syncthreads();
    const float* xrow = xg + (size_t)t * B_ * (4 * HD_);
#pragma unroll
    for (int j = 0; j < 8; j++) {
      const int e  = tid * 8 + j;        // e = b*Hd + hd
      const int b  = e >> 8;
      const int hd = e & (HD_ - 1);
      const float* xb = xrow + (size_t)b * (4 * HD_);
      const float* gb = g    + (size_t)b * (4 * HD_);
      float gi = xb[hd]            + gb[hd];
      float gf = xb[HD_ + hd]      + gb[HD_ + hd];
      float gz = xb[2 * HD_ + hd]  + gb[2 * HD_ + hd];
      float go = xb[3 * HD_ + hd]  + gb[3 * HD_ + hd];
      float cv = sigm(gf) * c[j] + sigm(gi) * tanhf(gz);
      float hv = sigm(go) * tanhf(cv);
      c[j] = cv;
      hbuf[e] = (h16)hv;
      xnext[((size_t)t * B_ + b) * H_ + d * HD_ + hd] = (h16)hv;
      if (s == T_ - 1) { fh[foff + e] = hv; fc[foff + e] = cv; }
    }
    __syncthreads();
  }
}

// ---------------- decoder LSTM scan (one WG per layer) ---------------------
__global__ __launch_bounds__(1024) void dec_scan_k(
    const float* __restrict__ yg,
    const h16* __restrict__ whh,
    const float* __restrict__ h0, const float* __restrict__ c0,
    h16* __restrict__ ynext,
    float* __restrict__ g)
{
  __shared__ __align__(16) h16 hbuf[B_ * H_];    // 32 KB
  const int tid  = threadIdx.x;
  const int lane = tid & 31;
  const int wave = tid >> 5;

  float c[16];
#pragma unroll
  for (int j = 0; j < 16; j++) {
    const int e = tid * 16 + j;
    c[j] = c0[e];
    hbuf[e] = (h16)h0[e];
  }
  __syncthreads();

  for (int s = 0; s < T_; s++) {
    for (int tile = wave; tile < 256; tile += 32) {
      const int m0 = (tile & 1) * 16;
      const int n0 = (tile >> 1) * 16;
      v8f acc = {};
      for (int k0 = 0; k0 < H_; k0 += 32) {
        v16h a = load_a16(hbuf, H_, m0, k0, lane);
        v16h b = load_b16(whh,  H_, n0, k0, lane);
        acc = WMMA_F32_F16(a, b, acc);
      }
      const int n  = n0 + (lane & 15);
      const int hh = lane >> 4;
#pragma unroll
      for (int r = 0; r < 8; r++)
        g[(size_t)(m0 + 8 * hh + r) * (4 * H_) + n] = acc[r];
    }
    __threadfence();
    __syncthreads();
    const float* yrow = yg + (size_t)s * B_ * (4 * H_);
#pragma unroll
    for (int j = 0; j < 16; j++) {
      const int e  = tid * 16 + j;       // e = b*H + h
      const int b  = e >> 9;
      const int hh = e & (H_ - 1);
      const float* xb = yrow + (size_t)b * (4 * H_);
      const float* gb = g    + (size_t)b * (4 * H_);
      float gi = xb[hh]           + gb[hh];
      float gf = xb[H_ + hh]      + gb[H_ + hh];
      float gz = xb[2 * H_ + hh]  + gb[2 * H_ + hh];
      float go = xb[3 * H_ + hh]  + gb[3 * H_ + hh];
      float cv = sigm(gf) * c[j] + sigm(gi) * tanhf(gz);
      float hv = sigm(go) * tanhf(cv);
      c[j] = cv;
      hbuf[e] = (h16)hv;
      ynext[((size_t)s * B_ + b) * H_ + hh] = (h16)hv;
    }
    __syncthreads();
  }
}

// ---------------- host orchestration ---------------------------------------
extern "C" void kernel_launch(void* const* d_in, const int* in_sizes, int n_in,
                              void* d_out, int out_size, void* d_ws, size_t ws_size,
                              hipStream_t stream) {
  (void)in_sizes; (void)n_in; (void)out_size; (void)ws_size;
  const int*   src_tok  = (const int*)  d_in[0];
  const int*   prev_tok = (const int*)  d_in[1];
  const float* W_emb    = (const float*)d_in[2];
  const float* b_emb    = (const float*)d_in[3];
  const float* enc_Wih  = (const float*)d_in[4];
  const float* enc_Whh  = (const float*)d_in[5];
  const float* enc_bih  = (const float*)d_in[6];
  const float* enc_bhh  = (const float*)d_in[7];
  const float* dec_Wih  = (const float*)d_in[8];
  const float* dec_Whh  = (const float*)d_in[9];
  const float* dec_bih  = (const float*)d_in[10];
  const float* dec_bhh  = (const float*)d_in[11];
  const float* W1       = (const float*)d_in[12];
  const float* b1       = (const float*)d_in[13];
  const float* W2       = (const float*)d_in[14];
  const float* b2       = (const float*)d_in[15];
  float* logits = (float*)d_out;

  char* p = (char*)d_ws;
  auto alloc = [&](size_t bytes) -> void* {
    void* r = (void*)p;
    p += (bytes + 255) & ~(size_t)255;
    return r;
  };
  const size_t S_act = (size_t)BT_ * H_;
  h16* encWih_h = (h16*)alloc(2ull*2*1024*512 * 2);
  h16* encWhh_h = (h16*)alloc(2ull*2*1024*256 * 2);
  h16* decWih_h = (h16*)alloc(2ull*2048*512   * 2);
  h16* decWhh_h = (h16*)alloc(2ull*2048*512   * 2);
  h16* W1_h     = (h16*)alloc(512ull*1024     * 2);
  h16* W2_h     = (h16*)alloc((size_t)V_*H_   * 2);
  h16* src_h    = (h16*)alloc(S_act * 2);
  h16* prev_h   = (h16*)alloc(S_act * 2);
  h16* xa_h     = (h16*)alloc(S_act * 2);
  h16* xb_h     = (h16*)alloc(S_act * 2);
  h16* outT_h   = (h16*)alloc(S_act * 2);
  h16* ya_h     = (h16*)alloc(S_act * 2);
  h16* yb_h     = (h16*)alloc(S_act * 2);
  h16* att_h    = (h16*)alloc((size_t)B_*T_*T_ * 2);
  h16* info_h   = (h16*)alloc((size_t)BT_*2*H_ * 2);
  h16* hmid_h   = (h16*)alloc(S_act * 2);
  float* xg0    = (float*)alloc((size_t)BT_*4*HD_ * 4);
  float* xg1    = (float*)alloc((size_t)BT_*4*HD_ * 4);
  float* yg     = (float*)alloc((size_t)BT_*4*H_  * 4);
  float* gsc    = (float*)alloc(2ull*B_*4*HD_     * 4);
  float* scores = (float*)alloc((size_t)B_*T_*T_  * 4);
  float* fh     = (float*)alloc(2ull*2*B_*HD_     * 4);
  float* fc     = (float*)alloc(2ull*2*B_*HD_     * 4);

  auto gemmT = [&](const h16* A, long lda, long sA, const h16* Bt, long ldb, long sB,
                   float* C, long ldc, long sC, h16* Ch, long ldch, long sCh,
                   const float* bias0, const float* bias1,
                   int M, int N, int K, int relu, int batch) {
    dim3 g((unsigned)(N / 128), (unsigned)(M / 128), (unsigned)batch);
    gemm_t128<<<g, 256, 0, stream>>>(A, lda, sA, Bt, ldb, sB, C, ldc, sC,
                                     Ch, ldch, sCh, bias0, bias1, K, relu);
  };
  auto gemm = [&](const h16* A, long lda, long sA, const h16* Bt, long ldb, long sB,
                  float* C, long ldc, long sC, h16* Ch, long ldch, long sCh,
                  const float* bias0, const float* bias1,
                  int M, int N, int K, int relu, int batch) {
    dim3 g((unsigned)(N / 16), (unsigned)(M / 16), (unsigned)batch);
    gemm16<<<g, 32, 0, stream>>>(A, lda, sA, Bt, ldb, sB, C, ldc, sC,
                                 Ch, ldch, sCh, bias0, bias1, K, relu);
  };
  auto cvt = [&](const float* s, h16* d, long n) {
    int blocks = (int)((n + 255) / 256);
    if (blocks > 4096) blocks = 4096;
    f32_to_f16_k<<<blocks, 256, 0, stream>>>(s, d, n);
  };

  cvt(enc_Wih, encWih_h, 2L*2*1024*512);
  cvt(enc_Whh, encWhh_h, 2L*2*1024*256);
  cvt(dec_Wih, decWih_h, 2L*2048*512);
  cvt(dec_Whh, decWhh_h, 2L*2048*512);
  cvt(W1, W1_h, 512L*1024);
  cvt(W2, W2_h, (long)V_*H_);

  {
    long n = (long)BT_ * H_;
    embed_k<<<(unsigned)((n + 255) / 256), 256, 0, stream>>>(src_tok,  W_emb, b_emb, src_h);
    embed_k<<<(unsigned)((n + 255) / 256), 256, 0, stream>>>(prev_tok, W_emb, b_emb, prev_h);
  }

  // ---- encoder ----
  const h16* xin = src_h;
  h16* xbufs[2] = { xa_h, xb_h };
  for (int l = 0; l < L_; l++) {
    for (int d = 0; d < 2; d++) {
      gemmT(xin, H_, 0,
            encWih_h + ((size_t)(l*2 + d)) * 1024 * 512, H_, 0,
            d ? xg1 : xg0, 4*HD_, 0, nullptr, 0, 0,
            enc_bih + (size_t)(l*2 + d) * 1024, enc_bhh + (size_t)(l*2 + d) * 1024,
            BT_, 4*HD_, H_, 0, 1);
    }
    enc_scan_k<<<2, 1024, 0, stream>>>(xg0, xg1,
        encWhh_h + (size_t)l * 2 * 1024 * 256, xbufs[l], gsc, fh, fc, l);
    xin = xbufs[l];
  }
  const h16* out_h = xin;

  // ---- decoder ----
  const h16* yin = prev_h;
  h16* ybufs[2] = { ya_h, yb_h };
  for (int l = 0; l < L_; l++) {
    gemmT(yin, H_, 0,
          decWih_h + (size_t)l * 2048 * 512, H_, 0,
          yg, 4*H_, 0, nullptr, 0, 0,
          dec_bih + (size_t)l * 2048, dec_bhh + (size_t)l * 2048,
          BT_, 4*H_, H_, 0, 1);
    dec_scan_k<<<1, 1024, 0, stream>>>(yg,
        decWhh_h + (size_t)l * 2048 * 512,
        fh + (size_t)l * 2 * B_ * HD_, fc + (size_t)l * 2 * B_ * HD_,
        ybufs[l], gsc);
    yin = ybufs[l];
  }
  const h16* fo_h = yin;

  // ---- attention ----
  {
    long n = (long)B_ * H_ * T_;
    transpose_k<<<(unsigned)((n + 255) / 256), 256, 0, stream>>>(out_h, outT_h);
  }
  gemmT(fo_h, (long)B_*H_, H_, out_h, (long)B_*H_, H_,
        scores, T_, (long)T_*T_, nullptr, 0, 0, nullptr, nullptr,
        T_, T_, H_, 0, B_);
  softmax_k<<<B_*T_, 128, 0, stream>>>(scores, att_h);
  gemmT(att_h, T_, (long)T_*T_, outT_h, T_, (long)H_*T_,
        nullptr, 0, 0, info_h, 2*H_, (long)T_*2*H_, nullptr, nullptr,
        T_, H_, T_, 0, B_);
  {
    long n = (long)BT_ * H_;
    concat_fo_k<<<(unsigned)((n + 255) / 256), 256, 0, stream>>>(fo_h, info_h);
  }

  // ---- head ----
  gemmT(info_h, 2*H_, 0, W1_h, 2*H_, 0,
        nullptr, 0, 0, hmid_h, H_, 0, b1, nullptr,
        BT_, H_, 2*H_, 1, 1);
  gemmT(hmid_h, H_, 0, W2_h, H_, 0,
        logits, V_, 0, nullptr, 0, 0, b2, nullptr,
        BT_, 9984, H_, 0, 1);
  gemm(hmid_h, H_, 0, W2_h + 9984L * H_, H_, 0,
       logits + 9984, V_, 0, nullptr, 0, 0, b2 + 9984, nullptr,
       BT_, 16, H_, 0, 1);
}
